// GTrXLCell_82343112999507
// MI455X (gfx1250) — matrix operand order repaired and checked by reading, not compiled, not run on hardware
//
#include <hip/hip_runtime.h>
#include <math.h>

typedef __attribute__((ext_vector_type(16))) __bf16 v16bf;
typedef __attribute__((ext_vector_type(4)))  __bf16 v4bf;
typedef __attribute__((ext_vector_type(8)))  float  v8f;

#define BB      4096
#define DIN     1024
#define DMODEL  2048
#define DFF     8192

#define BLK_M 128
#define BLK_N 128
#define BLK_K 64
#define LDSS  (BLK_K + 8)   // padded LDS row stride (bf16 elems) to dodge bank conflicts

enum { EP_F32 = 0, EP_BF16 = 1, EP_GATE = 2, EP_GELU = 3, EP_RESID_DUAL = 4 };

// C(M,N) = A(M,K) @ W(N,K)^T + bias, computed in bf16 WMMA with fp32 accumulate.
// W is always fp32 in global memory and converted to bf16 while staging to LDS.
// A is either fp32 (converted in-flight) or already bf16 in workspace.
template <typename AT, int EP>
__global__ __launch_bounds__(256) void gemm_wmma_bf16(
    const AT*    __restrict__ A,
    const float* __restrict__ W,
    const float* __restrict__ bias,
    void*        __restrict__ out,
    const float* __restrict__ e0,   // GATE: h_prev | RESID_DUAL: h_hat
    const float* __restrict__ e1,   // GATE: gate_a
    const float* __restrict__ e2,   // GATE: gate_b
    int M, int N, int K)
{
  __shared__ alignas(16) __bf16 As[BLK_M * LDSS];
  __shared__ alignas(16) __bf16 Bs[BLK_N * LDSS];

  const int tid  = threadIdx.x;
  const int lane = tid & 31;
  const int wv   = tid >> 5;      // wave 0..7
  const int wm   = wv >> 1;       // 0..3 : 32-row group
  const int wn   = wv & 1;        // 0..1 : 64-col group
  const int hl   = lane >> 4;     // half-wave
  const int l16  = lane & 15;
  const int m0   = blockIdx.y * BLK_M;
  const int n0   = blockIdx.x * BLK_N;

  v8f acc[2][4];
#pragma unroll
  for (int mi = 0; mi < 2; ++mi)
#pragma unroll
    for (int ni = 0; ni < 4; ++ni)
#pragma unroll
      for (int i = 0; i < 8; ++i) acc[mi][ni][i] = 0.0f;

  for (int kb = 0; kb < K; kb += BLK_K) {
    // ---- stage A tile (128 x 64) into LDS as bf16 ----
    if constexpr (sizeof(AT) == 4) {          // fp32 source: convert in flight
#pragma unroll
      for (int i = 0; i < 8; ++i) {
        int id = tid + i * 256;               // 2048 float4 chunks
        int r  = id >> 4;
        int c  = (id & 15) << 2;
        float4 f = *(const float4*)(A + (size_t)(m0 + r) * K + kb + c);
        v4bf h;
        h[0] = (__bf16)f.x; h[1] = (__bf16)f.y; h[2] = (__bf16)f.z; h[3] = (__bf16)f.w;
        *(v4bf*)(&As[r * LDSS + c]) = h;
      }
    } else {                                  // bf16 source: straight 16B copies
#pragma unroll
      for (int i = 0; i < 4; ++i) {
        int id = tid + i * 256;               // 1024 16B chunks
        int r  = id >> 3;
        int c  = (id & 7) << 3;
        *(float4*)(&As[r * LDSS + c]) =
            *(const float4*)((const __bf16*)A + (size_t)(m0 + r) * K + kb + c);
      }
    }
    // ---- stage W tile (128 rows of N x 64 of K) into LDS as bf16 ----
#pragma unroll
    for (int i = 0; i < 8; ++i) {
      int id = tid + i * 256;
      int r  = id >> 4;
      int c  = (id & 15) << 2;
      float4 f = *(const float4*)(W + (size_t)(n0 + r) * K + kb + c);
      v4bf h;
      h[0] = (__bf16)f.x; h[1] = (__bf16)f.y; h[2] = (__bf16)f.z; h[3] = (__bf16)f.w;
      *(v4bf*)(&Bs[r * LDSS + c]) = h;
    }
    __syncthreads();

#pragma unroll
    for (int ks = 0; ks < BLK_K; ks += 32) {
      // A fragment (16x32 bf16): lane l16 = row M; half-wave splits K into 0..7 / 8..15
      v16bf afrag[2];
#pragma unroll
      for (int mi = 0; mi < 2; ++mi) {
        const __bf16* p = &As[(wm * 32 + mi * 16 + l16) * LDSS + ks];
        float4* d = (float4*)&afrag[mi];
        d[0] = *(const float4*)(p + hl * 8);
        d[1] = *(const float4*)(p + 16 + hl * 8);
      }
      // B fragment (32x16 bf16): lane l16 = col N; half-wave splits K into 0..7 / 8..15
      v16bf bfrag[4];
#pragma unroll
      for (int ni = 0; ni < 4; ++ni) {
        const __bf16* p = &Bs[(wn * 64 + ni * 16 + l16) * LDSS + ks];
        float4* d = (float4*)&bfrag[ni];
        d[0] = *(const float4*)(p + hl * 8);
        d[1] = *(const float4*)(p + 16 + hl * 8);
      }
#pragma unroll
      for (int mi = 0; mi < 2; ++mi)
#pragma unroll
        for (int ni = 0; ni < 4; ++ni)
          acc[mi][ni] = __builtin_amdgcn_wmma_f32_16x16x32_bf16(
              false, afrag[mi], false, bfrag[ni], (short)0, acc[mi][ni], false, false);
    }
    __syncthreads();
  }

  // ---- fused epilogue ----
  const size_t MN = (size_t)M * N;
#pragma unroll
  for (int mi = 0; mi < 2; ++mi) {
#pragma unroll
    for (int ni = 0; ni < 4; ++ni) {
      const int n  = n0 + wn * 64 + ni * 16 + l16;
      const int mb = m0 + wm * 32 + mi * 16 + hl * 8;
      const float bv = bias[n];
      float sa = 0.f, sb = 0.f;
      if constexpr (EP == EP_GATE) {
        sa = 1.0f / (1.0f + __expf(-e1[n]));
        sb = 1.0f / (1.0f + __expf(-e2[n]));
      }
#pragma unroll
      for (int r = 0; r < 8; ++r) {
        const size_t idx = (size_t)(mb + r) * N + n;
        float val = acc[mi][ni][r] + bv;
        if constexpr (EP == EP_F32) {
          ((float*)out)[idx] = val;
        } else if constexpr (EP == EP_BF16) {
          ((__bf16*)out)[idx] = (__bf16)val;
        } else if constexpr (EP == EP_GATE) {
          ((float*)out)[idx] = sa * e0[idx] + sb * val;   // h_hat
        } else if constexpr (EP == EP_GELU) {
          float g = 0.5f * val * (1.0f + erff(val * 0.70710678f));
          ((__bf16*)out)[idx] = (__bf16)g;
        } else {                                          // EP_RESID_DUAL
          float hnew = val + e0[idx];
          float* o = (float*)out;
          o[idx]      = hnew;
          o[idx + MN] = hnew;                             // tuple: (h_new, h_new)
        }
      }
    }
  }
}

// LayerNorm over D=2048 (one block per row), output bf16 for the next WMMA GEMM.
__global__ __launch_bounds__(256) void layernorm2048_bf16(
    const float* __restrict__ x, const float* __restrict__ g,
    const float* __restrict__ b, __bf16* __restrict__ o)
{
  const int row = blockIdx.x;
  const int tid = threadIdx.x;
  const float* xr = x + (size_t)row * DMODEL;
  float v[8], s = 0.f, s2 = 0.f;
#pragma unroll
  for (int i = 0; i < 8; ++i) {
    float f = xr[tid + i * 256];
    v[i] = f; s += f; s2 += f * f;
  }
#pragma unroll
  for (int off = 16; off > 0; off >>= 1) {
    s  += __shfl_xor(s,  off, 32);
    s2 += __shfl_xor(s2, off, 32);
  }
  __shared__ float ps[8], ps2[8];
  if ((tid & 31) == 0) { ps[tid >> 5] = s; ps2[tid >> 5] = s2; }
  __syncthreads();
  float ts = 0.f, ts2 = 0.f;
#pragma unroll
  for (int i = 0; i < 8; ++i) { ts += ps[i]; ts2 += ps2[i]; }
  const float mean = ts * (1.0f / DMODEL);
  const float var  = ts2 * (1.0f / DMODEL) - mean * mean;
  const float inv  = rsqrtf(var + 1e-5f);
#pragma unroll
  for (int i = 0; i < 8; ++i) {
    const int c = tid + i * 256;
    o[(size_t)row * DMODEL + c] = (__bf16)((v[i] - mean) * inv * g[c] + b[c]);
  }
}

extern "C" void kernel_launch(void* const* d_in, const int* in_sizes, int n_in,
                              void* d_out, int out_size, void* d_ws, size_t ws_size,
                              hipStream_t stream) {
  (void)in_sizes; (void)n_in; (void)out_size; (void)ws_size;

  const float* x_t      = (const float*)d_in[0];   // (4096,1024)
  const float* h_prev   = (const float*)d_in[1];   // (4096,2048)
  const float* W_proj   = (const float*)d_in[2];   // (2048,1024)
  const float* b_proj   = (const float*)d_in[3];
  const float* in_w     = (const float*)d_in[4];   // (6144,2048)
  const float* in_b     = (const float*)d_in[5];
  const float* out_w    = (const float*)d_in[6];   // (2048,2048)
  const float* out_b    = (const float*)d_in[7];
  const float* ln1_g    = (const float*)d_in[8];
  const float* ln1_b    = (const float*)d_in[9];
  const float* ln2_g    = (const float*)d_in[10];
  const float* ln2_b    = (const float*)d_in[11];
  const float* gate_a   = (const float*)d_in[12];
  const float* gate_b   = (const float*)d_in[13];
  const float* ffn_w1   = (const float*)d_in[14];  // (8192,2048)
  const float* ffn_b1   = (const float*)d_in[15];
  const float* ffn_w2   = (const float*)d_in[16];  // (2048,8192)
  const float* ffn_b2   = (const float*)d_in[17];

  // softmax over seq_len==1 is identically 1 -> attention output = V.
  // Only the V third of in_proj is needed; q/k GEMMs are dead code.
  const float* Wv = in_w + (size_t)2 * DMODEL * DMODEL;
  const float* bv = in_b + (size_t)2 * DMODEL;

  // workspace layout (bytes)
  char* ws = (char*)d_ws;
  float*  x_f32 = (float*) (ws + 0);           // 4096*2048 f32 = 32 MB
  float*  h_hat = (float*) (ws + 33554432);    // 32 MB
  __bf16* qn    = (__bf16*)(ws + 67108864);    // 16 MB
  __bf16* vbuf  = (__bf16*)(ws + 83886080);    // 16 MB
  __bf16* ybuf  = (__bf16*)(ws + 100663296);   // 16 MB
  __bf16* act   = (__bf16*)(ws + 117440512);   // 4096*8192 bf16 = 64 MB

  const dim3 blk(256);
  const dim3 gD (DMODEL / BLK_N, BB / BLK_M);  // (16,32)
  const dim3 gFF(DFF    / BLK_N, BB / BLK_M);  // (64,32)

  // 1) x = x_t @ W_proj^T + b_proj            (fp32 A, fp32 out)
  gemm_wmma_bf16<float, EP_F32><<<gD, blk, 0, stream>>>(
      x_t, W_proj, b_proj, x_f32, nullptr, nullptr, nullptr, BB, DMODEL, DIN);
  // 2) qn = LN1(x)  -> bf16
  layernorm2048_bf16<<<dim3(BB), blk, 0, stream>>>(x_f32, ln1_g, ln1_b, qn);
  // 3) v = qn @ Wv^T + bv  -> bf16
  gemm_wmma_bf16<__bf16, EP_BF16><<<gD, blk, 0, stream>>>(
      qn, Wv, bv, vbuf, nullptr, nullptr, nullptr, BB, DMODEL, DMODEL);
  // 4) h_hat = sig(ga)*h_prev + sig(gb)*(v @ out_w^T + out_b)  -> fp32
  gemm_wmma_bf16<__bf16, EP_GATE><<<gD, blk, 0, stream>>>(
      vbuf, out_w, out_b, h_hat, h_prev, gate_a, gate_b, BB, DMODEL, DMODEL);
  // 5) y = LN2(h_hat) -> bf16
  layernorm2048_bf16<<<dim3(BB), blk, 0, stream>>>(h_hat, ln2_g, ln2_b, ybuf);
  // 6) act = gelu(y @ ffn_w1^T + b1) -> bf16
  gemm_wmma_bf16<__bf16, EP_GELU><<<gFF, blk, 0, stream>>>(
      ybuf, ffn_w1, ffn_b1, act, nullptr, nullptr, nullptr, BB, DFF, DMODEL);
  // 7) h_new = h_hat + act @ ffn_w2^T + b2 -> write twice (tuple output)
  gemm_wmma_bf16<__bf16, EP_RESID_DUAL><<<gD, blk, 0, stream>>>(
      act, ffn_w2, ffn_b2, d_out, h_hat, nullptr, nullptr, BB, DMODEL, DFF);
}
